// Attention_54408645706407
// MI455X (gfx1250) — matrix operand order repaired and checked by reading, not compiled, not run on hardware
//
#include <hip/hip_runtime.h>
#include <hip/hip_bf16.h>

typedef __bf16 bf16;
typedef __attribute__((ext_vector_type(16))) __bf16 v16bf;
typedef __attribute__((ext_vector_type(4)))  __bf16 v4bf;
typedef __attribute__((ext_vector_type(8)))  float  v8f;
typedef __attribute__((ext_vector_type(4)))  unsigned int u32x4;
typedef __attribute__((ext_vector_type(4)))  unsigned int v4u;
typedef __attribute__((ext_vector_type(8)))  int v8i_;
typedef __attribute__((ext_vector_type(4)))  int v4i_;

#define DIMX   1024
#define HEADS  16
#define GROUPS 4
#define DHEAD  64
#define INNER  1024
#define KVD    512
#define BB     4
#define NN     2048
#define ROWS   (BB*NN)          // 8192
#define LOG2E  1.4426950408889634f

union AB { v16bf v; u32x4 u[2]; };

// ---- CDNA5 data-movement feature detection --------------------------------
#if defined(__has_builtin)
#  if __has_builtin(__builtin_amdgcn_global_load_async_to_lds_b128)
#    define HAS_ASYNC_LDS 1
#  endif
#  if __has_builtin(__builtin_amdgcn_tensor_load_to_lds)
#    define HAS_TDM 1
#  endif
#endif

#if defined(HAS_ASYNC_LDS)
#  if __has_builtin(__builtin_amdgcn_s_wait_asynccnt)
#    define WAIT_ASYNC() __builtin_amdgcn_s_wait_asynccnt(0)
#  else
#    define WAIT_ASYNC() asm volatile("s_wait_asynccnt 0x0" ::: "memory")
#  endif
#else
#  define WAIT_ASYNC() ((void)0)
#endif

// 16-byte global -> LDS copy: async DMA path if available, sync fallback else.
__device__ __forceinline__ void cp16(bf16* lds_dst, const bf16* gsrc) {
#ifdef HAS_ASYNC_LDS
    __builtin_amdgcn_global_load_async_to_lds_b128(
        (__attribute__((address_space(1))) v4i_*)gsrc,
        (__attribute__((address_space(3))) v4i_*)lds_dst,
        0, 0);
#else
    *(u32x4*)lds_dst = *(const u32x4*)gsrc;
#endif
}

__device__ __forceinline__ unsigned lds_byte_off(const bf16* p) {
    return (unsigned)(unsigned long)(const __attribute__((address_space(3))) bf16*)p;
}

#ifdef HAS_TDM
// Issue one TDM 2D tile load: tile (tile_d0 x tile_d1) from a row-major
// (tensor_d1 x tensor_d0) bf16 tensor with inner stride stride0 (elements).
__device__ __forceinline__ void tdm_load_2d(const bf16* gsrc, unsigned lds_off,
                                            unsigned tensor_d0, unsigned tensor_d1,
                                            unsigned tile_d0, unsigned tile_d1,
                                            unsigned stride0) {
    unsigned long ga = (unsigned long)gsrc;
    v4u g0;
    g0[0] = 1u;                                        // count=1 (valid user D#)
    g0[1] = lds_off;                                   // lds_addr (bytes)
    g0[2] = (unsigned)ga;                              // global_addr[31:0]
    g0[3] = (unsigned)((ga >> 32) & 0x01FFFFFFu) | 0x80000000u;  // addr[56:32] | type=2
    v8i_ g1;
    g1[0] = 1 << 16;                                   // data_size = 1 (2 bytes)
    g1[1] = (int)((tensor_d0 & 0xFFFFu) << 16);        // tensor_dim0[15:0]
    g1[2] = (int)((tensor_d0 >> 16) | ((tensor_d1 & 0xFFFFu) << 16));
    g1[3] = (int)((tensor_d1 >> 16) | (tile_d0 << 16));// tile_dim0
    g1[4] = (int)tile_d1;                              // tile_dim1 (tile_dim2 = 0)
    g1[5] = (int)stride0;                              // tensor_dim0_stride[31:0]
    g1[6] = 0;                                         // stride0[47:32], dim1_stride lo
    g1[7] = 0;
    v4i_ z4 = { 0, 0, 0, 0 };
#if __clang_major__ >= 23
    v8i_ z8 = { 0, 0, 0, 0, 0, 0, 0, 0 };
    __builtin_amdgcn_tensor_load_to_lds(g0, g1, z4, z4, z8, 0);
#else
    __builtin_amdgcn_tensor_load_to_lds(g0, g1, z4, z4, 0);
#endif
}
#endif

// ---------------------------------------------------------------------------
// Weight transpose + cast fp32 -> bf16.  dst[n*K + k] = src[k*N + n]
// ---------------------------------------------------------------------------
__global__ __launch_bounds__(256)
void k_transpose_cast(const float* __restrict__ W, bf16* __restrict__ Wt, int K, int N) {
    long id = (long)blockIdx.x * 256 + threadIdx.x;
    if (id >= (long)K * N) return;
    int n = (int)(id / K), k = (int)(id % K);
    Wt[id] = (bf16)W[(long)k * N + n];
}

// ---------------------------------------------------------------------------
// Row LayerNorm over DIMX=1024 + cast to bf16. One block (256 thr) per row.
// ---------------------------------------------------------------------------
__global__ __launch_bounds__(256)
void k_ln_cast(const float* __restrict__ x, const float* __restrict__ sc,
               const float* __restrict__ bi, bf16* __restrict__ xn) {
    int row = blockIdx.x, t = threadIdx.x;
    const float4* xp = (const float4*)(x + (long)row * DIMX);
    float4 v = xp[t];
    float s = v.x + v.y + v.z + v.w;
    float q = v.x*v.x + v.y*v.y + v.z*v.z + v.w*v.w;
    for (int m = 16; m >= 1; m >>= 1) { s += __shfl_xor(s, m, 32); q += __shfl_xor(q, m, 32); }
    __shared__ float ss[8], sq[8];
    int w = t >> 5, l = t & 31;
    if (l == 0) { ss[w] = s; sq[w] = q; }
    __syncthreads();
    float ts = 0.f, tq = 0.f;
    for (int i = 0; i < 8; i++) { ts += ss[i]; tq += sq[i]; }
    float mu = ts * (1.0f / DIMX);
    float var = tq * (1.0f / DIMX) - mu * mu;
    float rs = rsqrtf(var + 1e-6f);
    int c0 = t * 4;
    float vv[4] = { v.x, v.y, v.z, v.w };
    v4bf o;
    for (int e = 0; e < 4; e++) o[e] = (bf16)((vv[e] - mu) * rs * sc[c0 + e] + bi[c0 + e]);
    *(v4bf*)(xn + (long)row * DIMX + c0) = o;
}

// ---------------------------------------------------------------------------
// bf16 WMMA GEMM: C[M,N] = A[M,K] * Bt[N,K]^T.  128x128 workgroup tile,
// 8 waves (2x4), each wave 64x32 via 4x2 wmma accumulators, K-step 32.
// Async global->LDS double buffering.
// ---------------------------------------------------------------------------
__global__ __launch_bounds__(256)
void k_gemm_bf16(const bf16* __restrict__ A, const bf16* __restrict__ Bt,
                 void* __restrict__ Cout, int M, int N, int K, int storeF32) {
    __shared__ __align__(32) bf16 sA[2][128 * 32];
    __shared__ __align__(32) bf16 sB[2][128 * 32];
    int t = threadIdx.x, w = t >> 5, l = t & 31, lh = l & 15, hh = l >> 4;
    int wm = w & 1, wn = w >> 1;
    long m0 = (long)blockIdx.x * 128, n0 = (long)blockIdx.y * 128;

    v8f acc[4][2];
    for (int i = 0; i < 4; i++)
        for (int j = 0; j < 2; j++)
            for (int r = 0; r < 8; r++) acc[i][j][r] = 0.f;

    auto stage = [&](int buf, int k0) {
        for (int i = 0; i < 2; i++) {
            int idx = t + 256 * i, row = idx >> 2, kp = idx & 3;
            cp16(&sA[buf][row * 32 + kp * 8], A + (m0 + row) * K + k0 + kp * 8);
            cp16(&sB[buf][row * 32 + kp * 8], Bt + (n0 + row) * K + k0 + kp * 8);
        }
    };

    stage(0, 0);
    WAIT_ASYNC();
    __syncthreads();

    int cur = 0;
    for (int k0 = 0; k0 < K; k0 += 32) {
        if (k0 + 32 < K) stage(cur ^ 1, k0 + 32);   // overlap with compute

        AB a[4]; v16bf bmat[2];
        for (int mt = 0; mt < 4; mt++) {
            const bf16* p = &sA[cur][(wm * 64 + mt * 16 + lh) * 32];
            a[mt].u[0] = *(const u32x4*)(p + 16 * hh);
            a[mt].u[1] = *(const u32x4*)(p + 8 + 16 * hh);
        }
        for (int nt = 0; nt < 2; nt++)
            bmat[nt] = *(const v16bf*)(&sB[cur][(wn * 32 + nt * 16 + lh) * 32 + 16 * hh]);

        for (int mt = 0; mt < 4; mt++)
            for (int nt = 0; nt < 2; nt++)
                acc[mt][nt] = __builtin_amdgcn_wmma_f32_16x16x32_bf16(
                    false, a[mt].v, false, bmat[nt], (short)0, acc[mt][nt], false, false);

        WAIT_ASYNC();
        __syncthreads();
        cur ^= 1;
    }

    for (int mt = 0; mt < 4; mt++)
        for (int nt = 0; nt < 2; nt++)
            for (int r = 0; r < 8; r++) {
                long m = m0 + wm * 64 + mt * 16 + r + 8 * hh;
                long n = n0 + wn * 32 + nt * 16 + lh;
                float val = acc[mt][nt][r];
                if (storeF32) ((float*)Cout)[m * N + n] = val;
                else          ((bf16*)Cout)[m * N + n] = (bf16)val;
            }
}

// ---------------------------------------------------------------------------
// Per-head LN for q: one wave per head (d=64, 2 elems/lane), 16 waves/row.
// ---------------------------------------------------------------------------
__global__ __launch_bounds__(512)
void k_head_ln_q(const bf16* __restrict__ qraw, const float* __restrict__ sc,
                 const float* __restrict__ bi, bf16* __restrict__ qout) {
    int row = blockIdx.x;                 // b*N + n
    int b = row >> 11, n = row & 2047;
    int h = threadIdx.x >> 5, l = threadIdx.x & 31, d0 = l * 2;
    const bf16* p = qraw + (long)row * INNER + h * DHEAD + d0;
    float x0 = (float)p[0], x1 = (float)p[1];
    float s = x0 + x1, q = x0 * x0 + x1 * x1;
    for (int m = 16; m >= 1; m >>= 1) { s += __shfl_xor(s, m, 32); q += __shfl_xor(q, m, 32); }
    float mu = s * (1.f / 64), var = q * (1.f / 64) - mu * mu, rs = rsqrtf(var + 1e-6f);
    bf16* o = qout + ((long)(b * HEADS + h) * NN + n) * DHEAD + d0;
    o[0] = (bf16)((x0 - mu) * rs * sc[d0] + bi[d0]);
    o[1] = (bf16)((x1 - mu) * rs * sc[d0 + 1] + bi[d0 + 1]);
}

// ---------------------------------------------------------------------------
// Per-group LN for k (waves 0-3) + v transpose to (b,g,d,n) (waves 4-7).
// ---------------------------------------------------------------------------
__global__ __launch_bounds__(256)
void k_head_ln_kv(const bf16* __restrict__ kvraw, const float* __restrict__ sc,
                  const float* __restrict__ bi, bf16* __restrict__ kout,
                  bf16* __restrict__ vt) {
    int row = blockIdx.x;
    int b = row >> 11, n = row & 2047;
    int w = threadIdx.x >> 5, l = threadIdx.x & 31, d0 = l * 2;
    if (w < 4) {
        int g = w;
        const bf16* p = kvraw + (long)row * KVD + g * DHEAD + d0;
        float x0 = (float)p[0], x1 = (float)p[1];
        float s = x0 + x1, q = x0 * x0 + x1 * x1;
        for (int m = 16; m >= 1; m >>= 1) { s += __shfl_xor(s, m, 32); q += __shfl_xor(q, m, 32); }
        float mu = s * (1.f / 64), var = q * (1.f / 64) - mu * mu, rs = rsqrtf(var + 1e-6f);
        bf16* o = kout + ((long)(b * GROUPS + g) * NN + n) * DHEAD + d0;
        o[0] = (bf16)((x0 - mu) * rs * sc[d0] + bi[d0]);
        o[1] = (bf16)((x1 - mu) * rs * sc[d0 + 1] + bi[d0 + 1]);
    } else {
        int g = w - 4;
        const bf16* p = kvraw + (long)row * KVD + 256 + g * DHEAD + d0;
        vt[((long)(b * GROUPS + g) * DHEAD + d0) * NN + n]     = p[0];
        vt[((long)(b * GROUPS + g) * DHEAD + d0 + 1) * NN + n] = p[1];
    }
}

// ---------------------------------------------------------------------------
// Flash attention: block = (b,h) x 128-query slab, 8 waves x 16 rows each.
// K/Vt 64-key tiles staged by the Tensor Data Mover (double buffered);
// online softmax; P routed through per-wave LDS tile (C-layout -> A-layout).
// ---------------------------------------------------------------------------
__global__ __launch_bounds__(256)
void k_attn(const bf16* __restrict__ q, const bf16* __restrict__ k,
            const bf16* __restrict__ vt, bf16* __restrict__ out) {
    __shared__ __align__(32) bf16 sK[2][64 * 64];
    __shared__ __align__(32) bf16 sV[2][64 * 64];
    __shared__ __align__(32) bf16 sP[8][16 * 64];
    int bh = blockIdx.x;
    int b = bh >> 4, h = bh & 15, g = h >> 2;
    int n0 = blockIdx.y * 128;
    int t = threadIdx.x, w = t >> 5, l = t & 31, lh = l & 15, hh = l >> 4;

    const bf16* qbase = q  + (long)(b * HEADS  + h) * NN * DHEAD;
    const bf16* kbase = k  + (long)(b * GROUPS + g) * NN * DHEAD;
    const bf16* vbase = vt + (long)(b * GROUPS + g) * DHEAD * NN;

    AB qa[2];
    {
        const bf16* qp = qbase + (long)(n0 + w * 16 + lh) * DHEAD;
        for (int kk = 0; kk < 2; kk++) {
            qa[kk].u[0] = *(const u32x4*)(qp + kk * 32 + 16 * hh);
            qa[kk].u[1] = *(const u32x4*)(qp + kk * 32 + 8 + 16 * hh);
        }
    }

    float mrow[8], lrow[8];
    v8f o[4];
    for (int r = 0; r < 8; r++) { mrow[r] = -3.0e38f; lrow[r] = 0.f; }
    for (int dt = 0; dt < 4; dt++)
        for (int r = 0; r < 8; r++) o[dt][r] = 0.f;

    auto stage = [&](int buf, int j) {
#ifdef HAS_TDM
        if (w == 0) {
            // K tile: 64 keys x 64 d, contiguous rows of 64
            tdm_load_2d(kbase + (long)j * 64 * DHEAD, lds_byte_off(&sK[buf][0]),
                        DHEAD, NN, 64, 64, DHEAD);
            // Vt tile: 64 d-rows x 64 keys, row stride NN
            tdm_load_2d(vbase + j * 64, lds_byte_off(&sV[buf][0]),
                        NN, DHEAD, 64, 64, NN);
        }
#else
        for (int i = 0; i < 2; i++) {
            int idx = t + 256 * i, r0 = idx >> 3, cp = idx & 7;
            *(u32x4*)(&sK[buf][r0 * 64 + cp * 8]) =
                *(const u32x4*)(kbase + (long)(j * 64 + r0) * DHEAD + cp * 8);
            *(u32x4*)(&sV[buf][r0 * 64 + cp * 8]) =
                *(const u32x4*)(vbase + (long)r0 * NN + j * 64 + cp * 8);
        }
#endif
    };

    stage(0, 0);
    int cur = 0;
    for (int j = 0; j < NN / 64; j++) {
#ifdef HAS_TDM
        if (w == 0) __builtin_amdgcn_s_wait_tensorcnt(0);
#endif
        __syncthreads();                        // staged tile visible to all waves
        if (j + 1 < NN / 64) stage(cur ^ 1, j + 1);   // overlap next tile with compute

        // S = scale * Q Kt  (4 key-tiles of 16)
        v8f s[4];
        for (int nt = 0; nt < 4; nt++) {
            v8f c = { 0.f, 0.f, 0.f, 0.f, 0.f, 0.f, 0.f, 0.f };
            for (int kk = 0; kk < 2; kk++) {
                v16bf kb = *(const v16bf*)(&sK[cur][(nt * 16 + lh) * 64 + kk * 32 + 16 * hh]);
                c = __builtin_amdgcn_wmma_f32_16x16x32_bf16(
                        false, qa[kk].v, false, kb, (short)0, c, false, false);
            }
            for (int r = 0; r < 8; r++) s[nt][r] = c[r] * 0.125f;   // d^-0.5
        }

        // online softmax, row = r + 8*hh, cols striped over 16-lane halves
        for (int r = 0; r < 8; r++) {
            float mx = fmaxf(fmaxf(s[0][r], s[1][r]), fmaxf(s[2][r], s[3][r]));
            mx = fmaxf(mx, __shfl_xor(mx, 1, 32));
            mx = fmaxf(mx, __shfl_xor(mx, 2, 32));
            mx = fmaxf(mx, __shfl_xor(mx, 4, 32));
            mx = fmaxf(mx, __shfl_xor(mx, 8, 32));
            float mn = fmaxf(mrow[r], mx);
            float al = __builtin_exp2f((mrow[r] - mn) * LOG2E);
            float rsum = 0.f;
            for (int nt = 0; nt < 4; nt++) {
                float p = __builtin_exp2f((s[nt][r] - mn) * LOG2E);
                s[nt][r] = p; rsum += p;
            }
            rsum += __shfl_xor(rsum, 1, 32);
            rsum += __shfl_xor(rsum, 2, 32);
            rsum += __shfl_xor(rsum, 4, 32);
            rsum += __shfl_xor(rsum, 8, 32);
            lrow[r] = lrow[r] * al + rsum;
            mrow[r] = mn;
            for (int dt = 0; dt < 4; dt++) o[dt][r] *= al;
        }

        // P: C-layout -> per-wave LDS tile (16 x 64) -> A-layout
        bf16* pb = &sP[w][0];
        for (int nt = 0; nt < 4; nt++)
            for (int r = 0; r < 8; r++)
                pb[(r + 8 * hh) * 64 + nt * 16 + lh] = (bf16)s[nt][r];
        asm volatile("s_wait_dscnt 0x0" ::: "memory");   // intra-wave cross-lane LDS dep

        AB pa[2];
        for (int kk = 0; kk < 2; kk++) {
            const bf16* pp = pb + lh * 64 + kk * 32 + 16 * hh;
            pa[kk].u[0] = *(const u32x4*)(pp);
            pa[kk].u[1] = *(const u32x4*)(pp + 8);
        }

        // O += P V  (4 d-tiles of 16)
        for (int dt = 0; dt < 4; dt++)
            for (int kk = 0; kk < 2; kk++) {
                v16bf vb = *(const v16bf*)(&sV[cur][(dt * 16 + lh) * 64 + kk * 32 + 16 * hh]);
                o[dt] = __builtin_amdgcn_wmma_f32_16x16x32_bf16(
                            false, pa[kk].v, false, vb, (short)0, o[dt], false, false);
            }

        __syncthreads();                        // all reads of cur done before reuse
        cur ^= 1;
    }

    // epilogue: O / l -> attn_out (b, n, h*64 + d) bf16
    for (int r = 0; r < 8; r++) {
        float inv = 1.0f / lrow[r];
        long m = (long)b * NN + n0 + w * 16 + r + 8 * hh;
        for (int dt = 0; dt < 4; dt++)
            out[m * INNER + h * DHEAD + dt * 16 + lh] = (bf16)(o[dt][r] * inv);
    }
}

// ---------------------------------------------------------------------------
extern "C" void kernel_launch(void* const* d_in, const int* in_sizes, int n_in,
                              void* d_out, int out_size, void* d_ws, size_t ws_size,
                              hipStream_t stream) {
    const float* x    = (const float*)d_in[0];
    const float* lns  = (const float*)d_in[1];
    const float* lnb  = (const float*)d_in[2];
    const float* Wq   = (const float*)d_in[3];
    const float* Wkv  = (const float*)d_in[4];
    const float* qns  = (const float*)d_in[5];
    const float* qnb  = (const float*)d_in[6];
    const float* kns  = (const float*)d_in[7];
    const float* knb  = (const float*)d_in[8];
    const float* Wout = (const float*)d_in[9];

    char* ws = (char*)d_ws;
    size_t off = 0;
    auto alloc = [&](size_t bytes) {
        void* p = ws + off;
        off += (bytes + 255) & ~(size_t)255;
        return p;
    };
    bf16* xn    = (bf16*)alloc((size_t)ROWS * DIMX * 2);
    bf16* WqT   = (bf16*)alloc((size_t)INNER * DIMX * 2);
    bf16* WkvT  = (bf16*)alloc((size_t)KVD * DIMX * 2);
    bf16* WoT   = (bf16*)alloc((size_t)DIMX * INNER * 2);
    bf16* qraw  = (bf16*)alloc((size_t)ROWS * INNER * 2);
    bf16* kvraw = (bf16*)alloc((size_t)ROWS * KVD * 2);
    bf16* qh    = (bf16*)alloc((size_t)BB * HEADS * NN * DHEAD * 2);
    bf16* kh    = (bf16*)alloc((size_t)BB * GROUPS * NN * DHEAD * 2);
    bf16* vth   = (bf16*)alloc((size_t)BB * GROUPS * NN * DHEAD * 2);
    bf16* aout  = (bf16*)alloc((size_t)ROWS * INNER * 2);

    k_transpose_cast<<<(INNER * DIMX + 255) / 256, 256, 0, stream>>>(Wq,   WqT,  DIMX,  INNER);
    k_transpose_cast<<<(KVD   * DIMX + 255) / 256, 256, 0, stream>>>(Wkv,  WkvT, DIMX,  KVD);
    k_transpose_cast<<<(DIMX * INNER + 255) / 256, 256, 0, stream>>>(Wout, WoT,  INNER, DIMX);

    k_ln_cast<<<ROWS, 256, 0, stream>>>(x, lns, lnb, xn);

    k_gemm_bf16<<<dim3(ROWS / 128, INNER / 128), 256, 0, stream>>>(xn, WqT,  qraw,  ROWS, INNER, DIMX, 0);
    k_gemm_bf16<<<dim3(ROWS / 128, KVD   / 128), 256, 0, stream>>>(xn, WkvT, kvraw, ROWS, KVD,   DIMX, 0);

    k_head_ln_q <<<ROWS, 512, 0, stream>>>(qraw,  qns, qnb, qh);
    k_head_ln_kv<<<ROWS, 256, 0, stream>>>(kvraw, kns, knb, kh, vth);

    k_attn<<<dim3(BB * HEADS, NN / 128), 256, 0, stream>>>(qh, kh, vth, aout);

    k_gemm_bf16<<<dim3(ROWS / 128, DIMX / 128), 256, 0, stream>>>(aout, WoT, d_out, ROWS, DIMX, INNER, 1);
}
